// ECN_52656299049250
// MI455X (gfx1250) — matrix operand order repaired and checked by reading, not compiled, not run on hardware
//
#include <hip/hip_runtime.h>
#include <hip/hip_bf16.h>

typedef __attribute__((ext_vector_type(16))) _Float16 v16h;
typedef __attribute__((ext_vector_type(4)))  _Float16 v4h;
typedef __attribute__((ext_vector_type(8)))  float    v8f;

#define N_NODES 12500
#define N_EDGES 50000
#define N_GRAPH 500

// ---------------------------------------------------------------- utilities
__global__ void zero_kernel(float* p, int n) {
  int i = blockIdx.x * blockDim.x + threadIdx.x;
  for (; i < n; i += gridDim.x * blockDim.x) p[i] = 0.f;
}

__global__ void cvt_f16_kernel(const float* __restrict__ src, _Float16* __restrict__ dst, int n) {
  int i = blockIdx.x * blockDim.x + threadIdx.x;
  if (i < n) dst[i] = (_Float16)src[i];
}

// Reshape+transpose w2 [64, cin*cout] -> w2rT [cout, 64*cin] (f16),
// and b2 [cin*cout] -> b2rT [cout, cin] (f16).
__global__ void prep_w_kernel(const float* __restrict__ w2, const float* __restrict__ b2,
                              _Float16* __restrict__ w2rT, _Float16* __restrict__ b2rT,
                              int cin, int cout) {
  int idx = blockIdx.x * blockDim.x + threadIdx.x;
  int tot = 64 * cin * cout;
  if (idx < tot) {
    int o = idx / (64 * cin);
    int r = idx - o * (64 * cin);
    int k = r / cin, i = r - k * cin;
    w2rT[idx] = (_Float16)w2[k * (cin * cout) + i * cout + o];
  }
  if (idx < cin * cout) {
    int o = idx / cin, i = idx - o * cin;
    b2rT[idx] = (_Float16)b2[i * cout + o];
  }
}

// Hmid = relu(edge_attr @ w1 + b1), stored f16.  [E,16]x[16,64] -> [E,64]
__global__ void edge_hidden_kernel(const float* __restrict__ ea, const float* __restrict__ w1,
                                   const float* __restrict__ b1, _Float16* __restrict__ mid, int E) {
  int idx = blockIdx.x * blockDim.x + threadIdx.x;
  if (idx >= E * 64) return;
  int e = idx >> 6, j = idx & 63;
  const float* er = &ea[e * 16];
  float a = b1[j];
#pragma unroll
  for (int t = 0; t < 16; ++t) a += er[t] * w1[t * 64 + j];
  mid[idx] = (_Float16)(a > 0.f ? a : 0.f);
}

// agg = bias + h @ root   [N,cin]x[cin,cout]
__global__ void root_init_kernel(const float* __restrict__ h, const float* __restrict__ root,
                                 const float* __restrict__ bias, float* __restrict__ agg,
                                 int n, int cin, int cout) {
  int idx = blockIdx.x * blockDim.x + threadIdx.x;
  if (idx >= n * cout) return;
  int nn = idx / cout, o = idx - nn * cout;
  float a = bias[o];
  const float* hr = &h[nn * cin];
  for (int i = 0; i < cin; ++i) a += hr[i] * root[i * cout + o];
  agg[idx] = a;
}

// ------------------------------------------------- fused NNConv message GEMM
// msg[e,o] = sum_{k,i} Hmid[e,k] * h16[src[e],i] * w2[k,i,o] + sum_i h[i]*b2[i,o]
// Outer product P built in registers; each B fragment reused by MT edge tiles.
template <int CIN, int COUT, int MT>
__launch_bounds__((COUT / 16) * 32)
__global__ void nnconv_wmma_kernel(const _Float16* __restrict__ h16,
                                   const _Float16* __restrict__ mid,
                                   const _Float16* __restrict__ w2rT,
                                   const _Float16* __restrict__ b2rT,
                                   const int* __restrict__ srcs, const int* __restrict__ dsts,
                                   float* __restrict__ agg, int ntiles) {
  constexpr int NW = COUT / 16;
  constexpr int NTH = NW * 32;
  constexpr int KC = CIN / 32;
  __shared__ v4h s_h4[MT * 16 * CIN / 4];
  __shared__ v4h s_mid4[MT * 16 * 64 / 4];
  __shared__ int s_dst[MT][16];
  const _Float16* s_h = (const _Float16*)s_h4;
  const _Float16* s_mid = (const _Float16*)s_mid4;

  const int tile0 = blockIdx.x * MT;
  const int tid = threadIdx.x;

  // gather h[src] rows (8B vectors), clamped for the tail block
  {
    const v4h* hsrc = (const v4h*)h16;
    for (int idx = tid; idx < MT * 16 * (CIN / 4); idx += NTH) {
      int j = idx / (16 * (CIN / 4));
      int rem = idx - j * (16 * (CIN / 4));
      int r = rem / (CIN / 4), c4 = rem - r * (CIN / 4);
      int e = (tile0 + j) * 16 + r;
      if (e >= N_EDGES) e = N_EDGES - 1;
      s_h4[idx] = hsrc[srcs[e] * (CIN / 4) + c4];
    }
  }
  {
    const v4h* mid4 = (const v4h*)mid;
    for (int idx = tid; idx < MT * 16 * 16; idx += NTH) {  // 16 v4h per row of 64
      int j = idx / (16 * 16);
      int rc = idx - j * (16 * 16);
      int e16 = (tile0 + j) * 16;
      if (e16 > N_EDGES - 16) e16 = N_EDGES - 16;
      s_mid4[idx] = mid4[e16 * 16 + rc];
    }
  }
  if (tid < MT * 16) {
    int j = tid >> 4, r = tid & 15;
    int e = (tile0 + j) * 16 + r;
    if (e >= N_EDGES) e = N_EDGES - 1;
    s_dst[j][r] = dsts[e];
  }
  __syncthreads();

  const int lane = tid & 31;
  const int wave = tid >> 5;
  const int m = lane & 15;
  const int half = lane >> 4;
  const int half8 = half * 8;
  const int half16 = half * 16;
  const int ncol = wave * 16 + m;

  // Per-lane h_src rows held in contiguous k order, same as B:
// element e of chunk c holds h[c*32 + half*16 + e].
  v16h areg[MT][KC];
#pragma unroll
  for (int j = 0; j < MT; ++j)
#pragma unroll
    for (int c = 0; c < KC; ++c)
#pragma unroll
      for (int e = 0; e < 16; ++e)
        areg[j][c][e] = s_h[j * (16 * CIN) + m * CIN + c * 32 + (e + half16)];

  v8f acc[MT];
#pragma unroll
  for (int j = 0; j < MT; ++j) acc[j] = (v8f){};

  // bias term: h_src @ b2   (K = CIN)
#pragma unroll
  for (int c = 0; c < KC; ++c) {
    v16h b = *(const v16h*)&b2rT[ncol * CIN + c * 32 + half16];
#pragma unroll
    for (int j = 0; j < MT; ++j)
      acc[j] = __builtin_amdgcn_wmma_f32_16x16x32_f16(false, areg[j][c], false, b, (short)0,
                                                      acc[j], false, false);
  }

  // bilinear term: P @ w2r  (K = 64*CIN, chunked by 32); B reused across MT tiles
  const _Float16* wbase = &w2rT[ncol * (64 * CIN) + half16];
  for (int k = 0; k < 64; ++k) {
#pragma unroll
    for (int c = 0; c < KC; ++c) {
      v16h b = *(const v16h*)(wbase + k * CIN + c * 32);
#pragma unroll
      for (int j = 0; j < MT; ++j) {
        _Float16 mk = s_mid[j * (16 * 64) + m * 64 + k];
        v16h a;
#pragma unroll
        for (int e = 0; e < 16; ++e) a[e] = mk * areg[j][c][e];
        acc[j] = __builtin_amdgcn_wmma_f32_16x16x32_f16(false, a, false, b, (short)0,
                                                        acc[j], false, false);
      }
    }
  }

  // scatter-add into destination nodes (D layout: elem r -> row r + 8*half);
  // validity test is block-uniform, so EXEC was all-ones through the WMMA region.
#pragma unroll
  for (int j = 0; j < MT; ++j) {
    if (tile0 + j < ntiles) {
#pragma unroll
      for (int r = 0; r < 8; ++r)
        atomicAdd(&agg[s_dst[j][r + half8] * COUT + ncol], acc[j][r]);
    }
  }
}

// ------------------------------------------------------------- batch norm
__global__ void bn_stats_kernel(const float* __restrict__ h, float* __restrict__ stats,
                                int n, int cout) {
  __shared__ float ssum[64], ssq[64];
  int t = threadIdx.x;
  if (t < cout) { ssum[t] = 0.f; ssq[t] = 0.f; }
  __syncthreads();
  int total = n * cout;
  for (int idx = blockIdx.x * blockDim.x + t; idx < total; idx += gridDim.x * blockDim.x) {
    float v = h[idx];
    int ch = idx % cout;
    atomicAdd(&ssum[ch], v);
    atomicAdd(&ssq[ch], v * v);
  }
  __syncthreads();
  if (t < cout) {
    atomicAdd(&stats[t], ssum[t]);
    atomicAdd(&stats[cout + t], ssq[t]);
  }
}

__global__ void bn_apply_kernel(const float* __restrict__ agg, const float* __restrict__ stats,
                                const float* __restrict__ g, const float* __restrict__ b,
                                const float* __restrict__ res, float* __restrict__ hout,
                                _Float16* __restrict__ hout16, int n, int cout) {
  int idx = blockIdx.x * blockDim.x + threadIdx.x;
  if (idx >= n * cout) return;
  int ch = idx % cout;
  float mean = stats[ch] / (float)n;
  float var = stats[cout + ch] / (float)n - mean * mean;
  float v = (agg[idx] - mean) * rsqrtf(var + 1e-5f) * g[ch] + b[ch];
  v = v > 0.f ? v : 0.f;
  if (res) v += res[idx];
  hout[idx] = v;
  if (hout16) hout16[idx] = (_Float16)v;
}

// ------------------------------------------------------------- pooling + head
__global__ void pool_kernel(const float* __restrict__ h, const int* __restrict__ batch,
                            float* __restrict__ ssum, float* __restrict__ smax,
                            float* __restrict__ cnt, int n) {
  int idx = blockIdx.x * blockDim.x + threadIdx.x;
  if (idx >= n * 32) return;
  int nn = idx >> 5, ch = idx & 31;
  int g = batch[nn];
  float v = h[idx];
  atomicAdd(&ssum[g * 32 + ch], v);
  atomicMax((int*)&smax[g * 32 + ch], __float_as_int(v));  // v >= 0 after relu
  if (ch == 0) atomicAdd(&cnt[g], 1.f);
}

__global__ void head_kernel(const float* __restrict__ ssum, const float* __restrict__ smax,
                            const float* __restrict__ cnt,
                            const float* fc1w, const float* fc1b,
                            const float* fc2w, const float* fc2b,
                            const float* h0w, const float* h0b,
                            const float* h1w, const float* h1b,
                            const float* h2w, const float* h2b, float* __restrict__ out) {
  int g = blockIdx.x;
  __shared__ float z[96], z1[64], z2[64];
  int t = threadIdx.x;  // 64 threads
  if (t < 32) {
    float s = ssum[g * 32 + t];
    float c = cnt[g]; c = c > 1.f ? c : 1.f;
    z[t] = s / c;
    z[32 + t] = s;
    z[64 + t] = smax[g * 32 + t];
  }
  __syncthreads();
  float a = fc1b[t];
  for (int i = 0; i < 96; ++i) a += z[i] * fc1w[i * 64 + t];
  z1[t] = a > 0.f ? a : 0.f;
  __syncthreads();
  float b = fc2b[t];
  for (int i = 0; i < 64; ++i) b += z1[i] * fc2w[i * 64 + t];
  z2[t] = b > 0.f ? b : 0.f;
  __syncthreads();
  if (t < 4) {
    float v = h0b[t];
    for (int i = 0; i < 64; ++i) v += z2[i] * h0w[i * 4 + t];
    out[g * 4 + t] = v;
  } else if (t < 8) {
    int c = t - 4;
    float v = h1b[c];
    for (int i = 0; i < 64; ++i) v += z2[i] * h1w[i * 4 + c];
    out[2000 + g * 4 + c] = v;
  } else if (t < 10) {
    int c = t - 8;
    float v = h2b[c];
    for (int i = 0; i < 64; ++i) v += z2[i] * h2w[i * 2 + c];
    out[4000 + g * 2 + c] = v;
  }
}

// ---------------------------------------------------------------- driver
extern "C" void kernel_launch(void* const* d_in, const int* in_sizes, int n_in,
                              void* d_out, int out_size, void* d_ws, size_t ws_size,
                              hipStream_t stream) {
  const int N = N_NODES, E = N_EDGES, G = N_GRAPH;
  const float* x      = (const float*)d_in[0];
  const int*   ei     = (const int*)d_in[1];
  const float* ea     = (const float*)d_in[2];
  const int*   batch  = (const int*)d_in[3];
  // conv1
  const float* c1w1 = (const float*)d_in[4],  *c1b1 = (const float*)d_in[5];
  const float* c1w2 = (const float*)d_in[6],  *c1b2 = (const float*)d_in[7];
  const float* c1rt = (const float*)d_in[8],  *c1bs = (const float*)d_in[9];
  // conv2
  const float* c2w1 = (const float*)d_in[10], *c2b1 = (const float*)d_in[11];
  const float* c2w2 = (const float*)d_in[12], *c2b2 = (const float*)d_in[13];
  const float* c2rt = (const float*)d_in[14], *c2bs = (const float*)d_in[15];
  // conv3
  const float* c3w1 = (const float*)d_in[16], *c3b1 = (const float*)d_in[17];
  const float* c3w2 = (const float*)d_in[18], *c3b2 = (const float*)d_in[19];
  const float* c3rt = (const float*)d_in[20], *c3bs = (const float*)d_in[21];
  const float* bn1g = (const float*)d_in[22], *bn1b = (const float*)d_in[23];
  const float* bn2g = (const float*)d_in[24], *bn2b = (const float*)d_in[25];
  const float* bn3g = (const float*)d_in[26], *bn3b = (const float*)d_in[27];
  const float* fc1w = (const float*)d_in[28], *fc1b = (const float*)d_in[29];
  const float* fc2w = (const float*)d_in[30], *fc2b = (const float*)d_in[31];
  const float* h0w  = (const float*)d_in[32], *h0b  = (const float*)d_in[33];
  const float* h1w  = (const float*)d_in[34], *h1b  = (const float*)d_in[35];
  const float* h2w  = (const float*)d_in[36], *h2b  = (const float*)d_in[37];

  const int* srcs = ei;
  const int* dsts = ei + E;

  // workspace carve (256B aligned)
  char* base = (char*)d_ws;
  size_t off = 0;
  auto carve = [&](size_t bytes) { char* p = base + off; off += (bytes + 255) & ~(size_t)255; return p; };
  float*     hA    = (float*)carve((size_t)N * 64 * 4);
  float*     hB    = (float*)carve((size_t)N * 64 * 4);
  _Float16*  hf16  = (_Float16*)carve((size_t)N * 64 * 2);
  _Float16*  mid   = (_Float16*)carve((size_t)E * 64 * 2);
  float*     agg   = (float*)carve((size_t)N * 64 * 4);
  _Float16*  w2rT  = (_Float16*)carve((size_t)64 * 64 * 64 * 2);
  _Float16*  b2rT  = (_Float16*)carve((size_t)64 * 64 * 2);
  float*     stats = (float*)carve(128 * 4);
  float*     pools = (float*)carve((size_t)(G * 32 * 2 + G) * 4);
  float* ssum = pools;
  float* smax = pools + G * 32;
  float* cnt  = pools + G * 64;

  dim3 b256(256);
  const int ntiles = E / 16;                 // 3125
  const int MT = 4;
  const int mblocks = (ntiles + MT - 1) / MT;  // 782

  // ---------------- conv1 (32 -> 64) ----------------
  cvt_f16_kernel<<<(N * 32 + 255) / 256, b256, 0, stream>>>(x, hf16, N * 32);
  prep_w_kernel<<<(64 * 32 * 64 + 255) / 256, b256, 0, stream>>>(c1w2, c1b2, w2rT, b2rT, 32, 64);
  edge_hidden_kernel<<<(E * 64 + 255) / 256, b256, 0, stream>>>(ea, c1w1, c1b1, mid, E);
  root_init_kernel<<<(N * 64 + 255) / 256, b256, 0, stream>>>(x, c1rt, c1bs, agg, N, 32, 64);
  nnconv_wmma_kernel<32, 64, 4><<<mblocks, 128, 0, stream>>>(hf16, mid, w2rT, b2rT, srcs, dsts, agg, ntiles);
  zero_kernel<<<1, 128, 0, stream>>>(stats, 128);
  bn_stats_kernel<<<512, b256, 0, stream>>>(agg, stats, N, 64);
  bn_apply_kernel<<<(N * 64 + 255) / 256, b256, 0, stream>>>(agg, stats, bn1g, bn1b, nullptr, hA, hf16, N, 64);

  // ---------------- conv2 (64 -> 64), residual ----------------
  prep_w_kernel<<<(64 * 64 * 64 + 255) / 256, b256, 0, stream>>>(c2w2, c2b2, w2rT, b2rT, 64, 64);
  edge_hidden_kernel<<<(E * 64 + 255) / 256, b256, 0, stream>>>(ea, c2w1, c2b1, mid, E);
  root_init_kernel<<<(N * 64 + 255) / 256, b256, 0, stream>>>(hA, c2rt, c2bs, agg, N, 64, 64);
  nnconv_wmma_kernel<64, 64, 4><<<mblocks, 128, 0, stream>>>(hf16, mid, w2rT, b2rT, srcs, dsts, agg, ntiles);
  zero_kernel<<<1, 128, 0, stream>>>(stats, 128);
  bn_stats_kernel<<<512, b256, 0, stream>>>(agg, stats, N, 64);
  bn_apply_kernel<<<(N * 64 + 255) / 256, b256, 0, stream>>>(agg, stats, bn2g, bn2b, hA, hB, hf16, N, 64);

  // ---------------- conv3 (64 -> 32) ----------------
  prep_w_kernel<<<(64 * 64 * 32 + 255) / 256, b256, 0, stream>>>(c3w2, c3b2, w2rT, b2rT, 64, 32);
  edge_hidden_kernel<<<(E * 64 + 255) / 256, b256, 0, stream>>>(ea, c3w1, c3b1, mid, E);
  root_init_kernel<<<(N * 32 + 255) / 256, b256, 0, stream>>>(hB, c3rt, c3bs, agg, N, 64, 32);
  nnconv_wmma_kernel<64, 32, 4><<<mblocks, 64, 0, stream>>>(hf16, mid, w2rT, b2rT, srcs, dsts, agg, ntiles);
  zero_kernel<<<1, 128, 0, stream>>>(stats, 128);
  bn_stats_kernel<<<512, b256, 0, stream>>>(agg, stats, N, 32);
  bn_apply_kernel<<<(N * 32 + 255) / 256, b256, 0, stream>>>(agg, stats, bn3g, bn3b, nullptr, hA, nullptr, N, 32);

  // ---------------- pooling + MLP head ----------------
  zero_kernel<<<(G * 65 + 127) / 128, 128, 0, stream>>>(pools, G * 65);
  pool_kernel<<<(N * 32 + 255) / 256, b256, 0, stream>>>(hA, batch, ssum, smax, cnt, N);
  head_kernel<<<G, 64, 0, stream>>>(ssum, smax, cnt, fc1w, fc1b, fc2w, fc2b,
                                    h0w, h0b, h1w, h1b, h2w, h2b, (float*)d_out);
}
